// ScaledDotProductAttention_17789754540216
// MI455X (gfx1250) — matrix operand order repaired and checked by reading, not compiled, not run on hardware
//
#include <hip/hip_runtime.h>
#include <hip/hip_bf16.h>
#include <math.h>
#include <stdint.h>

// Problem sizes (fixed by the reference)
constexpr int BB = 8;
constexpr int QQ = 2048;
constexpr int KK = 2048;
constexpr int DD = 1024;
constexpr float SCALE = 0.03125f; // 1/sqrt(1024)

typedef __bf16 bf16_t;
typedef __attribute__((ext_vector_type(16))) __bf16 v16bf;
typedef __attribute__((ext_vector_type(8)))  __bf16 v8bf;
typedef __attribute__((ext_vector_type(4)))  __bf16 v4bf;
typedef __attribute__((ext_vector_type(8)))  float  v8f;

typedef unsigned int u32x4 __attribute__((ext_vector_type(4)));
typedef int          i32x4 __attribute__((ext_vector_type(4)));
typedef int          i32x8 __attribute__((ext_vector_type(8)));

#if defined(__has_builtin)
#if __has_builtin(__builtin_amdgcn_tensor_load_to_lds)
#define HAVE_TDM 1
#endif
#endif

// ---------------------------------------------------------------------------
// CDNA5 async global->LDS copies (ASYNCcnt) + waits.
// ---------------------------------------------------------------------------
__device__ __forceinline__ void async_b128(void* lds, const void* g) {
  uint32_t l = (uint32_t)(uintptr_t)lds;  // low 32 bits of generic ptr = LDS offset
  asm volatile("global_load_async_to_lds_b128 %0, %1, off"
               :: "v"(l), "v"(g) : "memory");
}
__device__ __forceinline__ void wait_async_le1() {
  asm volatile("s_wait_asynccnt 1" ::: "memory");
}
__device__ __forceinline__ void wait_async_le0() {
  asm volatile("s_wait_asynccnt 0" ::: "memory");
}

__device__ __forceinline__ void wait_tensor_le1() {
#if defined(__has_builtin) && __has_builtin(__builtin_amdgcn_s_wait_tensorcnt)
  __builtin_amdgcn_s_wait_tensorcnt((short)1);
#else
  asm volatile("s_wait_tensorcnt 1" ::: "memory");
#endif
}
__device__ __forceinline__ void wait_tensor_le0() {
#if defined(__has_builtin) && __has_builtin(__builtin_amdgcn_s_wait_tensorcnt)
  __builtin_amdgcn_s_wait_tensorcnt((short)0);
#else
  asm volatile("s_wait_tensorcnt 0" ::: "memory");
#endif
}

#ifdef HAVE_TDM
// ---------------------------------------------------------------------------
// TDM: one wave-uniform 2-D tile DMA, global -> LDS (TENSORcnt-tracked).
// D# per CDNA5 ISA:
//  group0: [1:0]=count(1), [63:32]=lds_addr, [120:64]=global_addr, [127:126]=2
//  group1: [17:16]=data_size(1 -> 2B), [79:48]=tensor_dim0, [111:80]=tensor_dim1,
//          [127:112]=tile_dim0, [143:128]=tile_dim1, [207:160]=tensor_dim0_stride
// tile = tile1 rows x tile0 elements, rows packed contiguously in LDS.
// ---------------------------------------------------------------------------
__device__ __forceinline__ void tdm_load_2d(uint32_t lds_addr, const void* gaddr,
                                            uint32_t tile0, uint32_t tile1,
                                            uint64_t tensor_dim0,
                                            uint64_t dim0_stride) {
  const uint64_t ga = (uint64_t)(uintptr_t)gaddr;
  union { uint64_t q[2]; u32x4 v; } g0;
  g0.q[0] = 1ull | ((uint64_t)lds_addr << 32);
  g0.q[1] = (ga & ((1ull << 57) - 1)) | (2ull << 62);
  union { uint64_t q[4]; i32x8 v; } g1;
  const uint64_t dim1 = 0x40000000ull;  // large (no OOB clipping of rows)
  g1.q[0] = (1ull << 16) | ((tensor_dim0 & 0xFFFFull) << 48);
  g1.q[1] = (tensor_dim0 >> 16) | (dim1 << 16) | ((uint64_t)tile0 << 48);
  g1.q[2] = (uint64_t)tile1 | ((dim0_stride & 0xFFFFFFFFull) << 32);
  g1.q[3] = (dim0_stride >> 32);
  i32x4 z4 = {};
#if __clang_major__ >= 23
  i32x8 z8 = {};
  __builtin_amdgcn_tensor_load_to_lds(g0.v, g1.v, z4, z4, z8, 0);
#else
  __builtin_amdgcn_tensor_load_to_lds(g0.v, g1.v, z4, z4, 0);
#endif
}
#endif  // HAVE_TDM

__device__ __forceinline__ v8f wmma_bf16(v16bf a, v16bf b, v8f c) {
  return __builtin_amdgcn_wmma_f32_16x16x32_bf16(false, a, false, b, (short)0, c,
                                                 false, false);
}

// A-frag (16x32 bf16) from fp32 LDS tile; `ar` = row base + hi*16.
__device__ __forceinline__ v16bf a_frag_from_lds(const float* ar) {
  v16bf a;
#pragma unroll
  for (int i = 0; i < 8; ++i) a[i] = (__bf16)ar[i];
#pragma unroll
  for (int i = 0; i < 8; ++i) a[8 + i] = (__bf16)ar[8 + i];
  return a;
}

// B-frag: lane's 16 contiguous bf16 (K-run).
__device__ __forceinline__ v16bf b_frag(const bf16_t* p) {
  v8bf lo = *(const v8bf*)p;
  v8bf hi = *(const v8bf*)(p + 8);
  return __builtin_shufflevector(lo, hi, 0, 1, 2, 3, 4, 5, 6, 7,
                                 8, 9, 10, 11, 12, 13, 14, 15);
}

// ---------------------------------------------------------------------------
// K conversion: Kbf[b,k,d] = bf16(K[b,k,d]) and KbfT[b,d,k] = bf16(K[b,k,d]).
// ---------------------------------------------------------------------------
__global__ void __launch_bounds__(256)
cvtKT_kernel(const float* __restrict__ Km, bf16_t* __restrict__ Kbf,
             bf16_t* __restrict__ KbfT) {
  __shared__ bf16_t tile[32][33];
  constexpr int KT = KK / 32, DT = DD / 32;
  int gb = blockIdx.x;
  int dt = gb % DT;
  int kt = (gb / DT) % KT;
  int b  = gb / (DT * KT);
  int t  = threadIdx.x;
  int lk = t >> 3;
  int ld = (t & 7) * 4;
  const float4 f =
      *(const float4*)(Km + ((size_t)b * KK + kt * 32 + lk) * DD + dt * 32 + ld);
  v4bf h = {(bf16_t)f.x, (bf16_t)f.y, (bf16_t)f.z, (bf16_t)f.w};
  tile[lk][ld + 0] = h[0];
  tile[lk][ld + 1] = h[1];
  tile[lk][ld + 2] = h[2];
  tile[lk][ld + 3] = h[3];
  *(v4bf*)(Kbf + ((size_t)b * KK + kt * 32 + lk) * DD + dt * 32 + ld) = h;
  __syncthreads();
  int ldr = t >> 3;
  int lkr = (t & 7) * 4;
  v4bf o = {tile[lkr + 0][ldr], tile[lkr + 1][ldr], tile[lkr + 2][ldr],
            tile[lkr + 3][ldr]};
  *(v4bf*)(KbfT + ((size_t)b * DD + dt * 32 + ldr) * KK + kt * 32 + lkr) = o;
}

// ---------------------------------------------------------------------------
// Unified GEMM: Out[b,m,n] = (SC ? SCALE : 1) * sum_r A[b,m,r] * Bm[b,n,r]
//   A : fp32 [BB, QQ, RED]; Bm : bf16 [BB, OUTW, RED]; Out : fp32 [BB, QQ, OUTW]
// 128 threads = 4 waves; wave -> 16(M) x 64(N).
//   A tile (16x32 fp32): async global->LDS, double-buffered (ASYNCcnt).
//   B panel (256 rows x 32 bf16 = 16 KB): ONE TDM load per block-iteration,
//   issued by wave 0, double-buffered (TENSORcnt). Fallback: direct b128 loads.
// ---------------------------------------------------------------------------
template <int RED, int OUTW, bool SC>
__global__ void __launch_bounds__(128)
gemm_bf_kernel(const float* __restrict__ A, const bf16_t* __restrict__ Bm,
               float* __restrict__ Out) {
  constexpr int WPB = 4;               // waves per block
  constexpr int MT = QQ / 16;
  constexpr int NS = OUTW / 64;
  constexpr int NB = NS / WPB;
  constexpr int ITERS = RED / 32;
  constexpr int PROWS = WPB * 64;      // 256 B-panel rows per block

  __shared__ float abuf[2][16 * 36];
#ifdef HAVE_TDM
  __shared__ bf16_t bbuf[2][PROWS * 32];
#endif

  const int tid  = threadIdx.x;
  const int lane = tid & 31;
  const int lm   = lane & 15;
  const int hi   = lane >> 4;
  const int w    = tid >> 5;

  int gb  = blockIdx.x;
  int nsB = gb % NB;
  int mt  = (gb / NB) % MT;
  int b   = gb / (NB * MT);
  int ns  = nsB * WPB + w;

  // A staging: 16 rows x 32 fp32 = 2 KB/stage; 16B per thread (async b128).
  const int srow = tid >> 3;
  const int scol = (tid & 7) * 4;
  const float* gsrc = A + ((size_t)b * QQ + mt * 16 + srow) * RED + scol;

#ifdef HAVE_TDM
  const bf16_t* gpanel = Bm + ((size_t)b * OUTW + (size_t)nsB * PROWS) * RED;
#else
  const bf16_t* kb = Bm + ((size_t)b * OUTW + ns * 64 + lm) * RED + hi * 16;
#endif

  v8f acc0 = {}, acc1 = {}, acc2 = {}, acc3 = {};

  async_b128(&abuf[0][srow * 36 + scol], gsrc);  // preload A stage 0
#ifdef HAVE_TDM
  if (w == 0)
    tdm_load_2d((uint32_t)(uintptr_t)&bbuf[0][0], gpanel, 32, PROWS,
                (uint64_t)RED, (uint64_t)RED);   // preload B stage 0
#endif

  for (int i = 0; i < ITERS; ++i) {
    const int d0 = i * 32;
    if (i + 1 < ITERS) {
      async_b128(&abuf[(i + 1) & 1][srow * 36 + scol], gsrc + d0 + 32);
#ifdef HAVE_TDM
      if (w == 0)
        tdm_load_2d((uint32_t)(uintptr_t)&bbuf[(i + 1) & 1][0],
                    gpanel + d0 + 32, 32, PROWS, (uint64_t)RED, (uint64_t)RED);
#endif
      wait_async_le1();
      wait_tensor_le1();
    } else {
      wait_async_le0();
      wait_tensor_le0();
    }
    __syncthreads();

    v16bf a = a_frag_from_lds(&abuf[i & 1][lm * 36 + hi * 16]);
#ifdef HAVE_TDM
    const bf16_t* bp = &bbuf[i & 1][((size_t)w * 64 + lm) * 32 + hi * 16];
    acc0 = wmma_bf16(a, b_frag(bp + 0 * 16 * 32), acc0);
    acc1 = wmma_bf16(a, b_frag(bp + 1 * 16 * 32), acc1);
    acc2 = wmma_bf16(a, b_frag(bp + 2 * 16 * 32), acc2);
    acc3 = wmma_bf16(a, b_frag(bp + 3 * 16 * 32), acc3);
#else
    const bf16_t* kp = kb + d0;
    __builtin_prefetch(kp + 64, 0, 1);
    acc0 = wmma_bf16(a, b_frag(kp + (size_t)0 * 16 * RED), acc0);
    acc1 = wmma_bf16(a, b_frag(kp + (size_t)1 * 16 * RED), acc1);
    acc2 = wmma_bf16(a, b_frag(kp + (size_t)2 * 16 * RED), acc2);
    acc3 = wmma_bf16(a, b_frag(kp + (size_t)3 * 16 * RED), acc3);
#endif
    __syncthreads();
  }

  // C/D layout: lanes 0-15: VGPR i -> M=i, N=lane; lanes 16-31: M=i+8.
  const size_t rowBase = (size_t)b * QQ + mt * 16;
#pragma unroll
  for (int i = 0; i < 8; ++i) {
    int m = i + hi * 8;
    size_t off = (rowBase + m) * (size_t)OUTW + ns * 64 + lm;
    if (SC) {
      Out[off + 0]  = acc0[i] * SCALE;
      Out[off + 16] = acc1[i] * SCALE;
      Out[off + 32] = acc2[i] * SCALE;
      Out[off + 48] = acc3[i] * SCALE;
    } else {
      Out[off + 0]  = acc0[i];
      Out[off + 16] = acc1[i];
      Out[off + 32] = acc2[i];
      Out[off + 48] = acc3[i];
    }
  }
}

// ---------------------------------------------------------------------------
// Row-wise softmax in place; one 256-thread block per row (2048 elements).
// ---------------------------------------------------------------------------
__global__ void __launch_bounds__(256)
softmax_kernel(float* __restrict__ attn) {
  __shared__ float red[8];
  const int row = blockIdx.x;
  float* p = attn + (size_t)row * KK;
  const int tid = threadIdx.x;

  float v[8];
  float m = -3.402823466e38f;
#pragma unroll
  for (int i = 0; i < 8; ++i) {
    v[i] = p[tid + i * 256];
    m = fmaxf(m, v[i]);
  }
#pragma unroll
  for (int off = 16; off > 0; off >>= 1) m = fmaxf(m, __shfl_xor(m, off, 32));
  if ((tid & 31) == 0) red[tid >> 5] = m;
  __syncthreads();
  float rowmax = red[0];
#pragma unroll
  for (int i = 1; i < 8; ++i) rowmax = fmaxf(rowmax, red[i]);
  __syncthreads();

  float s = 0.f;
#pragma unroll
  for (int i = 0; i < 8; ++i) {
    v[i] = __expf(v[i] - rowmax);
    s += v[i];
  }
#pragma unroll
  for (int off = 16; off > 0; off >>= 1) s += __shfl_xor(s, off, 32);
  if ((tid & 31) == 0) red[tid >> 5] = s;
  __syncthreads();
  float tot = 0.f;
#pragma unroll
  for (int i = 0; i < 8; ++i) tot += red[i];
  float inv = 1.0f / tot;
#pragma unroll
  for (int i = 0; i < 8; ++i) p[tid + i * 256] = v[i] * inv;
}

// ===========================================================================
// Fallback path (insufficient workspace): direct-fp32 kernels.
// ===========================================================================
__device__ __forceinline__ v16bf pack_a32(const float* __restrict__ p, int hi) {
  const float4* q0 = reinterpret_cast<const float4*>(p + hi * 16);
  const float4* q1 = reinterpret_cast<const float4*>(p + 8 + hi * 16);
  float4 f0 = q0[0], f1 = q0[1], f2 = q1[0], f3 = q1[1];
  float t[16] = {f0.x, f0.y, f0.z, f0.w, f1.x, f1.y, f1.z, f1.w,
                 f2.x, f2.y, f2.z, f2.w, f3.x, f3.y, f3.z, f3.w};
  v16bf r;
#pragma unroll
  for (int i = 0; i < 16; ++i) r[i] = (__bf16)t[i];
  return r;
}
__device__ __forceinline__ v16bf pack_b_contig(const float* __restrict__ p) {
  const float4* q = reinterpret_cast<const float4*>(p);
  float4 f0 = q[0], f1 = q[1], f2 = q[2], f3 = q[3];
  float t[16] = {f0.x, f0.y, f0.z, f0.w, f1.x, f1.y, f1.z, f1.w,
                 f2.x, f2.y, f2.z, f2.w, f3.x, f3.y, f3.z, f3.w};
  v16bf r;
#pragma unroll
  for (int i = 0; i < 16; ++i) r[i] = (__bf16)t[i];
  return r;
}
__device__ __forceinline__ v16bf pack_b_strided(const float* __restrict__ p,
                                                int strideFloats) {
  v16bf r;
#pragma unroll
  for (int i = 0; i < 16; ++i) r[i] = (__bf16)p[(size_t)i * strideFloats];
  return r;
}

__global__ void __launch_bounds__(256)
scores_fb_kernel(const float* __restrict__ Qm, const float* __restrict__ Km,
                 float* __restrict__ S) {
  constexpr int MT = QQ / 16, NS = KK / 64;
  const int wid  = blockIdx.x * (blockDim.x >> 5) + (threadIdx.x >> 5);
  const int lane = threadIdx.x & 31;
  const int lm = lane & 15, hi = lane >> 4;
  int b = wid / (MT * NS), r = wid % (MT * NS);
  int mt = r / NS, ns = r % NS;
  if (b >= BB) return;
  const float* qrow = Qm + ((size_t)b * QQ + mt * 16 + lm) * DD;
  const float* kb   = Km + ((size_t)b * KK + ns * 64) * DD;
  v8f acc0 = {}, acc1 = {}, acc2 = {}, acc3 = {};
  for (int d0 = 0; d0 < DD; d0 += 32) {
    v16bf a  = pack_a32(qrow + d0, hi);
    acc0 = wmma_bf16(a, pack_b_contig(kb + (size_t)(0 * 16 + lm) * DD + d0 + hi * 16), acc0);
    acc1 = wmma_bf16(a, pack_b_contig(kb + (size_t)(1 * 16 + lm) * DD + d0 + hi * 16), acc1);
    acc2 = wmma_bf16(a, pack_b_contig(kb + (size_t)(2 * 16 + lm) * DD + d0 + hi * 16), acc2);
    acc3 = wmma_bf16(a, pack_b_contig(kb + (size_t)(3 * 16 + lm) * DD + d0 + hi * 16), acc3);
  }
  const size_t rowBase = (size_t)b * QQ + mt * 16;
#pragma unroll
  for (int i = 0; i < 8; ++i) {
    int m = i + hi * 8;
    size_t off = (rowBase + m) * (size_t)KK + ns * 64 + lm;
    S[off + 0]  = acc0[i] * SCALE;
    S[off + 16] = acc1[i] * SCALE;
    S[off + 32] = acc2[i] * SCALE;
    S[off + 48] = acc3[i] * SCALE;
  }
}

__global__ void __launch_bounds__(256)
context_fb_kernel(const float* __restrict__ attn, const float* __restrict__ Km,
                  float* __restrict__ ctx) {
  constexpr int MT = QQ / 16, NS = DD / 64;
  const int wid  = blockIdx.x * (blockDim.x >> 5) + (threadIdx.x >> 5);
  const int lane = threadIdx.x & 31;
  const int lm = lane & 15, hi = lane >> 4;
  int b = wid / (MT * NS), r = wid % (MT * NS);
  int mt = r / NS, ns = r % NS;
  if (b >= BB) return;
  const float* arow = attn + ((size_t)b * QQ + mt * 16 + lm) * KK;
  const float* kb   = Km + (size_t)b * KK * DD;
  v8f acc0 = {}, acc1 = {}, acc2 = {}, acc3 = {};
  for (int k0 = 0; k0 < KK; k0 += 32) {
    v16bf a = pack_a32(arow + k0, hi);
    const float* bbase = kb + (size_t)(k0 + hi * 16) * DD + ns * 64 + lm;
    acc0 = wmma_bf16(a, pack_b_strided(bbase + 0,  DD), acc0);
    acc1 = wmma_bf16(a, pack_b_strided(bbase + 16, DD), acc1);
    acc2 = wmma_bf16(a, pack_b_strided(bbase + 32, DD), acc2);
    acc3 = wmma_bf16(a, pack_b_strided(bbase + 48, DD), acc3);
  }
  const size_t rowBase = (size_t)b * QQ + mt * 16;
#pragma unroll
  for (int i = 0; i < 8; ++i) {
    int m = i + hi * 8;
    size_t off = (rowBase + m) * (size_t)DD + ns * 64 + lm;
    ctx[off + 0]  = acc0[i];
    ctx[off + 16] = acc1[i];
    ctx[off + 32] = acc2[i];
    ctx[off + 48] = acc3[i];
  }
}

// ---------------------------------------------------------------------------
extern "C" void kernel_launch(void* const* d_in, const int* in_sizes, int n_in,
                              void* d_out, int out_size, void* d_ws, size_t ws_size,
                              hipStream_t stream) {
  const float* Qm = (const float*)d_in[0];
  const float* Km = (const float*)d_in[1];
  float* ctx  = (float*)d_out;               // [B,Q,D]
  float* attn = ctx + (size_t)BB * QQ * DD;  // [B,Q,K]

  const size_t kelems = (size_t)BB * KK * DD;
  const size_t need   = 2 * kelems * sizeof(bf16_t);  // Kbf + KbfT = 67 MB

  if (d_ws != nullptr && ws_size >= need) {
    bf16_t* Kbf  = (bf16_t*)d_ws;
    bf16_t* KbfT = Kbf + kelems;
    cvtKT_kernel<<<dim3(BB * (KK / 32) * (DD / 32)), dim3(256), 0, stream>>>(
        Km, Kbf, KbfT);
    gemm_bf_kernel<DD, KK, true>
        <<<dim3(BB * (QQ / 16) * (KK / 64) / 4), dim3(128), 0, stream>>>(
            Qm, Kbf, attn);
    softmax_kernel<<<dim3(BB * QQ), dim3(256), 0, stream>>>(attn);
    gemm_bf_kernel<KK, DD, false>
        <<<dim3(BB * (QQ / 16) * (DD / 64) / 4), dim3(128), 0, stream>>>(
            attn, KbfT, ctx);
  } else {
    int waves1 = BB * (QQ / 16) * (KK / 64);
    scores_fb_kernel<<<dim3(waves1 / 8), dim3(256), 0, stream>>>(Qm, Km, attn);
    softmax_kernel<<<dim3(BB * QQ), dim3(256), 0, stream>>>(attn);
    int waves2 = BB * (QQ / 16) * (DD / 64);
    context_fb_kernel<<<dim3(waves2 / 8), dim3(256), 0, stream>>>(attn, Km, ctx);
  }
}
